// CrossDomainFusion_7756710937370
// MI455X (gfx1250) — matrix operand order repaired and checked by reading, not compiled, not run on hardware
//
#include <hip/hip_runtime.h>
#include <hip/hip_bf16.h>
#include <math.h>

// ---------------- CDNA5 WMMA plumbing (gfx1250, wave32) ----------------
typedef __attribute__((ext_vector_type(16))) __bf16 v16bf;
typedef __attribute__((ext_vector_type(8)))  __bf16 v8bf;
typedef __attribute__((ext_vector_type(8)))  float  v8f;

union V16U { v16bf v; v8bf h[2]; };

// D = A(16x32) * B(32x16) + C, bf16 inputs, fp32 accumulate
__device__ __forceinline__ v8f wmma_bf16(v16bf a, v16bf b, v8f c) {
  return __builtin_amdgcn_wmma_f32_16x16x32_bf16(
      false, a, false, b, (short)0, c, false, false);
}

// A fragment (16x32 bf16). Source row-major [M,K], leading dim ld (elements).
// Per-lane layout (ISA 7.12.2): lane m (lo half) holds row m, K=kb..kb+7 and
// kb+16..kb+23; lane 16+m holds row m, K=kb+8..15 and kb+24..31.
// -> two contiguous 16B chunks at (kb + 8*hi) and (kb + 16 + 8*hi).
__device__ __forceinline__ v16bf fragA_row(const __bf16* base, int ld, int kb,
                                           int row, bool valid) {
  int hi = (threadIdx.x >> 4) & 1;
  V16U r;
  if (valid) {
    const __bf16* p = base + (size_t)row * ld + kb + 8 * hi;
    r.h[0] = *(const v8bf*)p;
    r.h[1] = *(const v8bf*)(p + 16);
  } else {
#pragma unroll
    for (int i = 0; i < 16; ++i) r.v[i] = (__bf16)0.0f;
  }
  return r.v;
}

__device__ __forceinline__ v16bf fragA(const __bf16* base, int ld, int kb) {
  return fragA_row(base, ld, kb, threadIdx.x & 15, true);
}

// B fragment (32x16 bf16). Source stored transposed: row-major [N,K], ld = K
// stride. Lane n (lo half) holds col n, K=kb..kb+7 and kb+16..kb+23; lane 16+n
// holds K=kb+8..15 and kb+24..31 -> two 16B chunks (same order as A).
__device__ __forceinline__ v16bf fragB(const __bf16* base, int ld, int kb) {
  int lane = threadIdx.x & 31;
  int n = lane & 15, hi = lane >> 4;
  const __bf16* p = base + (size_t)n * ld + kb + 8 * hi;
  V16U r;
  r.h[0] = *(const v8bf*)p;
  r.h[1] = *(const v8bf*)(p + 16);
  return r.v;
}

// ---------------- Problem constants ----------------
#define NB   8
#define TIN  1024
#define SEQ  2048
#define TD   256
#define SD   192
#define HD   512

// ---------------- Small prep kernels ----------------
__global__ __launch_bounds__(256) void cvt_time_k(const float* __restrict__ x,
                                                  __bf16* __restrict__ y, int n) {
  int i = blockIdx.x * 256 + threadIdx.x;
  if (i < n) y[i] = (__bf16)x[i];
}

// spec [B,3,64,2048] -> bf16 [B,2048,192]
__global__ __launch_bounds__(256) void cvt_spec_k(const float* __restrict__ sp,
                                                  __bf16* __restrict__ y) {
  int i = blockIdx.x * 256 + threadIdx.x;
  const int total = NB * SEQ * SD;
  if (i >= total) return;
  int d = i % SD;
  int s = (i / SD) % SEQ;
  int b = i / (SD * SEQ);
  y[i] = (__bf16)sp[((size_t)b * SD + d) * SEQ + s];
}

// spec_w [192,512] -> bf16 transposed [512,192]
__global__ __launch_bounds__(256) void cvt_sw_k(const float* __restrict__ w,
                                                __bf16* __restrict__ wT) {
  int i = blockIdx.x * 256 + threadIdx.x;
  if (i >= SD * HD) return;
  int k = i / HD, n = i % HD;
  wT[(size_t)n * SD + k] = (__bf16)w[i];
}

// WcT[k][n][cin] = bf16( sum_c conv_w[cin][c][k] * time_w[c][n] )
__global__ __launch_bounds__(256) void fold_w_k(const float* __restrict__ cw,
                                                const float* __restrict__ tw,
                                                __bf16* __restrict__ WcT) {
  int idx = blockIdx.x * 256 + threadIdx.x;
  if (idx >= 4 * HD * TD) return;
  int cin = idx & 255;
  int n   = (idx >> 8) & 511;
  int k   = idx >> 17;
  float s = 0.f;
  for (int c = 0; c < TD; ++c)
    s += cw[((size_t)cin * TD + c) * 4 + k] * tw[(size_t)c * HD + n];
  WcT[((size_t)k * HD + n) * TD + cin] = (__bf16)s;
}

__global__ __launch_bounds__(256) void fold_bias_k(const float* __restrict__ cb,
                                                   const float* __restrict__ tw,
                                                   const float* __restrict__ tb,
                                                   float* __restrict__ biasT) {
  int n = blockIdx.x * 256 + threadIdx.x;
  if (n >= HD) return;
  float s = tb[n];
  for (int c = 0; c < TD; ++c) s += cb[c] * tw[(size_t)c * HD + n];
  biasT[n] = s;
}

// ---------------- H_time: fused convT + projection (WMMA GEMM) ----------------
// Even t=2u:  H = x[u]@Wc1 + x[u-1]@Wc3 + bias
// Odd  t=2u+1: H = x[u+1]@Wc0 + x[u]@Wc2 + bias
// A fragments (both shifted-row tiles) are hoisted to registers and reused
// across the 4 output-column tiles handled by this wave.
__global__ __launch_bounds__(256) void htime_k(const __bf16* __restrict__ xbf,
                                               const __bf16* __restrict__ WcT,
                                               const float* __restrict__ biasT,
                                               __bf16* __restrict__ ht,
                                               __bf16* __restrict__ htT) {
  const int u0 = blockIdx.x * 16, par = blockIdx.y, b = blockIdx.z;
  const int lane = threadIdx.x & 31, w = threadIdx.x >> 5;
  const int m = lane & 15, hi = lane >> 4;
  const __bf16* xb = xbf + (size_t)b * TIN * TD;
  const __bf16* WA = WcT + (size_t)(1 - par) * HD * TD;
  const __bf16* WB = WcT + (size_t)(3 - par) * HD * TD;
  const int rA = u0 + par + m;       // shift = par
  const int rB = u0 + par - 1 + m;   // shift = par-1
  const bool vA = (unsigned)rA < (unsigned)TIN;
  const bool vB = (unsigned)rB < (unsigned)TIN;

  v16bf a0[TD / 32], a1[TD / 32];
#pragma unroll
  for (int k = 0; k < TD; k += 32) {
    a0[k >> 5] = fragA_row(xb, TD, k, rA, vA);
    a1[k >> 5] = fragA_row(xb, TD, k, rB, vB);
  }

  for (int i = 0; i < 4; ++i) {
    const int n0 = (w + 8 * i) * 16;
    v8f acc = {};
#pragma unroll
    for (int k = 0; k < TD; k += 32) {
      v16bf b0 = fragB(WA + (size_t)n0 * TD, TD, k);
      acc = wmma_bf16(a0[k >> 5], b0, acc);
      v16bf b1 = fragB(WB + (size_t)n0 * TD, TD, k);
      acc = wmma_bf16(a1[k >> 5], b1, acc);
    }
    const float bias = biasT[n0 + m];
#pragma unroll
    for (int v = 0; v < 8; ++v) {
      int t = 2 * (u0 + v + 8 * hi) + par;
      __bf16 bv = (__bf16)(acc[v] + bias);
      ht [((size_t)b * SEQ + t) * HD + n0 + m] = bv;
      htT[(size_t)(b * HD + n0 + m) * SEQ + t] = bv;
    }
  }
}

// ---------------- H_spec projection (WMMA GEMM) ----------------
__global__ __launch_bounds__(256) void hspec_k(const __bf16* __restrict__ specbf,
                                               const __bf16* __restrict__ swT,
                                               const float* __restrict__ sb,
                                               __bf16* __restrict__ hs,
                                               __bf16* __restrict__ hsT) {
  const int s0 = blockIdx.x * 16, b = blockIdx.z;
  const int lane = threadIdx.x & 31, w = threadIdx.x >> 5;
  const int m = lane & 15, hi = lane >> 4;
  const __bf16* A = specbf + (size_t)b * SEQ * SD;

  v16bf af[SD / 32];
#pragma unroll
  for (int k = 0; k < SD; k += 32) af[k >> 5] = fragA(A + (size_t)s0 * SD, SD, k);

  for (int i = 0; i < 4; ++i) {
    const int n0 = (w + 8 * i) * 16;
    v8f acc = {};
#pragma unroll
    for (int k = 0; k < SD; k += 32) {
      v16bf bb = fragB(swT + (size_t)n0 * SD, SD, k);
      acc = wmma_bf16(af[k >> 5], bb, acc);
    }
    const float bias = sb[n0 + m];
#pragma unroll
    for (int v = 0; v < 8; ++v) {
      int s = s0 + v + 8 * hi;
      __bf16 bv = (__bf16)(acc[v] + bias);
      hs [((size_t)b * SEQ + s) * HD + n0 + m] = bv;
      hsT[(size_t)(b * HD + n0 + m) * SEQ + s] = bv;
    }
  }
}

// ---------------- Flash cross-attention (one direction per launch) ----------
// Q row-major [SEQ,HD]; K row-major [SEQ,HD] (is the B operand for S=Q@K^T);
// VT = V transposed [HD,SEQ] (B operand for P@V). 8 waves / workgroup:
// 16 queries x 128-key blocks; S and P tiles staged in LDS; online softmax.
// Q fragments are register-resident for the whole kernel (16 x v16bf).
__global__ __launch_bounds__(256) void attn_k(const __bf16* __restrict__ Qg,
                                              const __bf16* __restrict__ Kg,
                                              const __bf16* __restrict__ VTg,
                                              float* __restrict__ outg,
                                              int colOff) {
  __shared__ float  sS[16][132];                      // padded: 4-bank row skew
  __shared__ __attribute__((aligned(16))) __bf16 sP[16][136];
  __shared__ float  sM[16], sL[16], sC[16];

  const int tid = threadIdx.x;
  const int lane = tid & 31, w = tid >> 5;
  const int m = lane & 15, hi = lane >> 4;
  const int b = blockIdx.y;
  const int q0 = blockIdx.x * 16;
  const __bf16* Q  = Qg  + (size_t)b * SEQ * HD;
  const __bf16* K  = Kg  + (size_t)b * SEQ * HD;
  const __bf16* VT = VTg + (size_t)b * HD * SEQ;
  const float scale = 0.04419417382f;                 // 1/sqrt(512)

  if (tid < 16) { sM[tid] = -1e30f; sL[tid] = 0.f; }

  // Q tile register-resident: shared by every key block (removes 15/16 of the
  // Q-side global traffic at ~128 VGPRs).
  v16bf qa[HD / 32];
#pragma unroll
  for (int k = 0; k < HD; k += 32) qa[k >> 5] = fragA(Q + (size_t)q0 * HD, HD, k);

  v8f acc[4];
#pragma unroll
  for (int t = 0; t < 4; ++t)
#pragma unroll
    for (int v = 0; v < 8; ++v) acc[t][v] = 0.f;

  for (int kb = 0; kb < SEQ; kb += 128) {
    // ---- S tile: wave w computes S[0:16][16w:16w+16] over K=512 ----
    {
      const __bf16* Krow = K + (size_t)(kb + 16 * w) * HD;
      if (kb + 128 < SEQ)                      // keep L0 ahead of the WMMA stream
        __builtin_prefetch((const void*)(Krow + 128 * HD), 0, 0);
      v8f s = {};
#pragma unroll
      for (int k = 0; k < HD; k += 32) {
        v16bf bb = fragB(Krow, HD, k);
        s = wmma_bf16(qa[k >> 5], bb, s);
      }
#pragma unroll
      for (int v = 0; v < 8; ++v) sS[v + 8 * hi][16 * w + m] = s[v] * scale;
    }
    __syncthreads();

    // ---- online softmax: 16 threads per row ----
    {
      const int row = tid >> 4, j0 = (tid & 15) * 8;
      float mx = -1e30f;
#pragma unroll
      for (int j = 0; j < 8; ++j) mx = fmaxf(mx, sS[row][j0 + j]);
#pragma unroll
      for (int d = 1; d < 16; d <<= 1) mx = fmaxf(mx, __shfl_xor(mx, d, 32));
      const float mold = sM[row];
      const float mnew = fmaxf(mold, mx);
      const float corr = __expf(mold - mnew);
      float ls = 0.f;
#pragma unroll
      for (int j = 0; j < 8; ++j) {
        float pv = __expf(sS[row][j0 + j] - mnew);
        sP[row][j0 + j] = (__bf16)pv;
        ls += pv;
      }
#pragma unroll
      for (int d = 1; d < 16; d <<= 1) ls += __shfl_xor(ls, d, 32);
      if ((tid & 15) == 0) {
        sM[row] = mnew;
        sL[row] = sL[row] * corr + ls;
        sC[row] = corr;
      }
    }
    __syncthreads();

    // ---- rescale accumulators, then P@V (A from LDS, B from VT) ----
    {
#pragma unroll
      for (int v = 0; v < 8; ++v) {
        float cf = sC[v + 8 * hi];
#pragma unroll
        for (int t = 0; t < 4; ++t) acc[t][v] *= cf;
      }
      v16bf pa[4];
#pragma unroll
      for (int kk = 0; kk < 4; ++kk) pa[kk] = fragA(&sP[0][0], 136, kk * 32);
#pragma unroll
      for (int t = 0; t < 4; ++t) {
        const __bf16* Vrow = VT + (size_t)(64 * w + 16 * t) * SEQ + kb;
        if (kb + 128 < SEQ)
          __builtin_prefetch((const void*)(Vrow + 128), 0, 0);
#pragma unroll
        for (int kk = 0; kk < 4; ++kk) {
          v16bf bb = fragB(Vrow, SEQ, kk * 32);
          acc[t] = wmma_bf16(pa[kk], bb, acc[t]);
        }
      }
    }
    // next-iteration __syncthreads() (after S write) guards sP/sC reuse
  }

  // ---- epilogue: divide by l, write fp32 output slice ----
#pragma unroll
  for (int v = 0; v < 8; ++v) {
    const int t = q0 + v + 8 * hi;
    const float inv = 1.0f / sL[v + 8 * hi];
    const size_t base = ((size_t)b * SEQ + t) * 1024 + colOff + 64 * w;
#pragma unroll
    for (int tt = 0; tt < 4; ++tt) outg[base + 16 * tt + m] = acc[tt][v] * inv;
  }
}

// ---------------- host side ----------------
extern "C" void kernel_launch(void* const* d_in, const int* in_sizes, int n_in,
                              void* d_out, int out_size, void* d_ws, size_t ws_size,
                              hipStream_t stream) {
  const float* tf     = (const float*)d_in[0];   // [8,1024,256]
  const float* spec   = (const float*)d_in[1];   // [8,3,64,2048]
  const float* conv_w = (const float*)d_in[2];   // [256,256,4]
  const float* conv_b = (const float*)d_in[3];   // [256]
  const float* time_w = (const float*)d_in[4];   // [256,512]
  const float* time_b = (const float*)d_in[5];   // [512]
  const float* spec_w = (const float*)d_in[6];   // [192,512]
  const float* spec_b = (const float*)d_in[7];   // [512]
  float* out = (float*)d_out;                    // [8,2048,1024]

  char* ws = (char*)d_ws;
  size_t off = 0;
  auto carve = [&](size_t bytes) -> char* {
    char* p = ws + off;
    off = (off + bytes + 255) & ~(size_t)255;
    return p;
  };
  __bf16* xbf    = (__bf16*)carve((size_t)NB * TIN * TD * 2);
  __bf16* specbf = (__bf16*)carve((size_t)NB * SEQ * SD * 2);
  __bf16* WcT    = (__bf16*)carve((size_t)4 * HD * TD * 2);
  __bf16* swT    = (__bf16*)carve((size_t)HD * SD * 2);
  float*  biasT  = (float*) carve((size_t)HD * 4);
  __bf16* ht     = (__bf16*)carve((size_t)NB * SEQ * HD * 2);
  __bf16* htT    = (__bf16*)carve((size_t)NB * HD * SEQ * 2);
  __bf16* hs     = (__bf16*)carve((size_t)NB * SEQ * HD * 2);
  __bf16* hsT    = (__bf16*)carve((size_t)NB * HD * SEQ * 2);

  {
    int n = NB * TIN * TD;
    cvt_time_k<<<(n + 255) / 256, 256, 0, stream>>>(tf, xbf, n);
  }
  cvt_spec_k<<<(NB * SEQ * SD + 255) / 256, 256, 0, stream>>>(spec, specbf);
  cvt_sw_k<<<(SD * HD + 255) / 256, 256, 0, stream>>>(spec_w, swT);
  fold_w_k<<<(4 * HD * TD + 255) / 256, 256, 0, stream>>>(conv_w, time_w, WcT);
  fold_bias_k<<<2, 256, 0, stream>>>(conv_b, time_w, time_b, biasT);

  htime_k<<<dim3(TIN / 16, 2, NB), 256, 0, stream>>>(xbf, WcT, biasT, ht, htT);
  hspec_k<<<dim3(SEQ / 16, 1, NB), 256, 0, stream>>>(specbf, swT, spec_b, hs, hsT);

  // fused_time = softmax(Ht Hs^T) Hs  -> out[:, :, 0:512]
  attn_k<<<dim3(SEQ / 16, NB), 256, 0, stream>>>(ht, hs, hsT, out, 0);
  // fused_spec = softmax(Hs Ht^T) Ht  -> out[:, :, 512:1024]
  attn_k<<<dim3(SEQ / 16, NB), 256, 0, stream>>>(hs, ht, htT, out, 512);
}
